// HGHypContrastiveLoss_40054865002718
// MI455X (gfx1250) — matrix operand order repaired and checked by reading, not compiled, not run on hardware
//
#include <hip/hip_runtime.h>
#include <hip/hip_bf16.h>

// ---- CDNA5 vector types (trivial, union-safe) ----
typedef __attribute__((ext_vector_type(16))) __bf16 v16bf;
typedef __attribute__((ext_vector_type(8)))  float  v8f;
typedef __attribute__((ext_vector_type(4)))  unsigned int u32x4;

union FragBF { u32x4 q[2]; v16bf v; };

#define N_TOT 8192
#define D_DIM 128
#define N_SPLIT 16          // column splits per 16-row tile
#define COLS_PER_SPLIT (N_TOT / N_SPLIT)   // 512 columns -> 32 tiles of 16

__device__ __forceinline__ float warp_sum32(float x) {
#pragma unroll
  for (int m = 1; m < 32; m <<= 1) x += __shfl_xor(x, m, 32);
  return x;
}

// ------------------------------------------------------------------
// Kernel 1: row-normalize, bf16 hi/lo split, n2 per row.
// ------------------------------------------------------------------
__global__ void hyp_normalize(const float* __restrict__ x,
                              __bf16* __restrict__ hi,
                              __bf16* __restrict__ lo,
                              float* __restrict__ n2) {
  const int lane = threadIdx.x & 31;
  const int row  = blockIdx.x * (blockDim.x >> 5) + (threadIdx.x >> 5);
  const float4 v = *(const float4*)(x + row * D_DIM + lane * 4);
  float ss = v.x * v.x + v.y * v.y + v.z * v.z + v.w * v.w;
  ss = warp_sum32(ss);
  const float rinv = 1.0f / fmaxf(sqrtf(ss), 1e-12f);
  float f[4] = {v.x * rinv, v.y * rinv, v.z * rinv, v.w * rinv};
  float s2 = f[0] * f[0] + f[1] * f[1] + f[2] * f[2] + f[3] * f[3];
  s2 = warp_sum32(s2);
  if (lane == 0) n2[row] = s2;
#pragma unroll
  for (int i = 0; i < 4; ++i) {
    __bf16 h = (__bf16)f[i];
    __bf16 l = (__bf16)(f[i] - (float)h);
    hi[row * D_DIM + lane * 4 + i] = h;
    lo[row * D_DIM + lane * 4 + i] = l;
  }
}

// ------------------------------------------------------------------
// Kernel 2: main pass. One wave = (16-row tile, 512-column slice).
// Gram tile: 3-term bf16 split on THREE independent WMMA accumulator
// chains (no WMMA->WMMA RAW hazards). adc <= 0 always, so exp-sums are
// accumulated unshifted; true row max tracked separately.
// ------------------------------------------------------------------
__global__ void __launch_bounds__(256)
hyp_main(const __bf16* __restrict__ hi, const __bf16* __restrict__ lo,
         const float* __restrict__ n2,
         const int* __restrict__ pl, const int* __restrict__ sl,
         float4* __restrict__ partials) {
  const int lane    = threadIdx.x & 31;
  const int wave    = blockIdx.x * (blockDim.x >> 5) + (threadIdx.x >> 5);
  const int rowTile = wave >> 4;       // 512 row tiles
  const int split   = wave & (N_SPLIT - 1);
  const int rowBase = rowTile << 4;
  const int half    = lane >> 4;       // 0: lanes 0-15, 1: lanes 16-31
  const int l15     = lane & 15;

  // ---- Preload A fragments (16x32 bf16 per K-chunk), hi and lo planes.
  const __bf16* arow_hi = hi + (size_t)(rowBase + l15) * D_DIM;
  const __bf16* arow_lo = lo + (size_t)(rowBase + l15) * D_DIM;
  FragBF a_hi[4], a_lo[4];
#pragma unroll
  for (int c = 0; c < 4; ++c) {
    a_hi[c].q[0] = *(const u32x4*)(arow_hi + c * 32 + half * 8);
    a_hi[c].q[1] = *(const u32x4*)(arow_hi + c * 32 + 16 + half * 8);
    a_lo[c].q[0] = *(const u32x4*)(arow_lo + c * 32 + half * 8);
    a_lo[c].q[1] = *(const u32x4*)(arow_lo + c * 32 + 16 + half * 8);
  }

  const float C1 = 0.05f, C2 = 0.1f, CC = 0.0025f;          // c, 2c, c^2
  const float SQRT_C = 0.22360679774997896f;
  const float ADC_LOG2 = -6.19969683f;                       // -1/(sqrt_c*T) * ln2
  const float EXP_K    = -8.944271909999159f;                // -1/(sqrt_c*T)

  // ---- Per-row, loop-invariant values for this lane's 8 accumulator slots.
  int   pli[8], sli[8];
  float y2[8], oneCy[8], dlane[8];
#pragma unroll
  for (int v = 0; v < 8; ++v) {
    const int r = rowBase + half * 8 + v;
    pli[v] = pl[r]; sli[v] = sl[r]; y2[v] = n2[r];
    oneCy[v] = 1.f + C1 * y2[v];
    dlane[v] = (l15 == half * 8 + v) ? 1.f : 0.f;   // diagonal lane pattern
  }

  // ---- Running state: row max, unshifted exp-sum, sum pos adc, pos count.
  float m_[8], S_[8], P_[8], np_[8];
#pragma unroll
  for (int v = 0; v < 8; ++v) { m_[v] = -INFINITY; S_[v] = 0.f; P_[v] = 0.f; np_[v] = 0.f; }

  for (int t = 0; t < COLS_PER_SPLIT / 16; ++t) {
    const int colBase = split * COLS_PER_SPLIT + t * 16;
    const int n = colBase + l15;                             // this lane's column
    const __bf16* bh = hi + (size_t)n * D_DIM;
    const __bf16* bl = lo + (size_t)n * D_DIM;

    // Speculative prefetch of next tile's B rows (OOB prefetch is dropped).
    __builtin_prefetch(hi + (size_t)(n + 16) * D_DIM, 0, 1);
    __builtin_prefetch(lo + (size_t)(n + 16) * D_DIM, 0, 1);

    // Three independent accumulator chains -> no WMMA->WMMA RAW stalls.
    v8f accA = {}, accB = {}, accC = {};
#pragma unroll
    for (int c = 0; c < 4; ++c) {
      FragBF bhF, blF;
      bhF.q[0] = *(const u32x4*)(bh + c * 32 + half * 8);
      bhF.q[1] = *(const u32x4*)(bh + c * 32 + 16 + half * 8);
      blF.q[0] = *(const u32x4*)(bl + c * 32 + half * 8);
      blF.q[1] = *(const u32x4*)(bl + c * 32 + 16 + half * 8);
      accA = __builtin_amdgcn_wmma_f32_16x16x32_bf16(false, a_hi[c].v, false, bhF.v,
                                                     (short)0, accA, false, false);
      accB = __builtin_amdgcn_wmma_f32_16x16x32_bf16(false, a_hi[c].v, false, blF.v,
                                                     (short)0, accB, false, false);
      accC = __builtin_amdgcn_wmma_f32_16x16x32_bf16(false, a_lo[c].v, false, bhF.v,
                                                     (short)0, accC, false, false);
    }
    const v8f acc = (accA + accB) + accC;

    // ---- Per-tile lane-uniform / scalar invariants.
    const float x2   = n2[n];
    const float Bq   = 1.f - C1 * x2;
    const float Bq2  = Bq * Bq;
    const float twoBq = 2.f * Bq;
    const float ccx2 = CC * x2;
    const float dtile = (colBase == rowBase) ? 1.f : 0.f;    // diag only in this tile
    const int   plj = pl[n], slj = sl[n];

#pragma unroll
    for (int v = 0; v < 8; ++v) {
      const float s  = acc[v];
      const float t0 = fmaf(-C2, s, 1.f);                    // 1 + 2c*xy
      const float Aq  = fmaf(C1, y2[v], t0);
      const float den = fmaxf(fmaf(ccx2, y2[v], t0), 1e-15f);
      const float u   = Aq * (-s);                           // A*xy
      const float nsq = fmaxf(fmaf(x2 * Aq, Aq, fmaf(twoBq, u, Bq2 * y2[v])), 0.f);
      float nd = SQRT_C * __builtin_amdgcn_sqrtf(nsq) * __builtin_amdgcn_rcpf(den);
      nd = fminf(nd, 1.f - 1e-7f);                           // nd >= 0 by construction
      const float ratio = (1.f + nd) * __builtin_amdgcn_rcpf(1.f - nd);
      const float lr  = __builtin_amdgcn_logf(ratio);        // log2(ratio) >= 0
      const float adc = ADC_LOG2 * lr;                       // <= 0
      const float e   = __builtin_amdgcn_exp2f(EXP_K * lr);  // exp(adc), no overflow

      // branch/compare-free diagonal mask; label mask via bitwise-or predicate
      const float ndiag = fmaf(-dtile, dlane[v], 1.f);       // 0 on diag, else 1
      const float sameF = ((plj == pli[v]) | (slj == sli[v])) ? 1.f : 0.f;
      const float posf  = sameF * ndiag;

      m_[v]  = fmaxf(m_[v], adc);        // diag included in max
      S_[v]  = fmaf(ndiag, e, S_[v]);    // diag excluded from sums
      P_[v]  = fmaf(posf, adc, P_[v]);
      np_[v] += posf;
    }
  }

  // ---- Merge across the 16 lanes of each half-wave (plain max/sum merges).
#pragma unroll
  for (int v = 0; v < 8; ++v) {
#pragma unroll
    for (int msk = 1; msk < 16; msk <<= 1) {
      m_[v]  = fmaxf(m_[v], __shfl_xor(m_[v], msk, 32));
      S_[v]  += __shfl_xor(S_[v], msk, 32);
      P_[v]  += __shfl_xor(P_[v], msk, 32);
      np_[v] += __shfl_xor(np_[v], msk, 32);
    }
    if (l15 == 0) {
      const int r = rowBase + half * 8 + v;
      partials[(size_t)r * N_SPLIT + split] = make_float4(m_[v], S_[v], P_[v], np_[v]);
    }
  }
}

// ------------------------------------------------------------------
// Kernel 3: merge partials per row, reduce to scalar loss.
// den = S0 * exp(-m) + 1e-8  ==  sum(exp(adc - m)) + 1e-8  (reference)
// ------------------------------------------------------------------
__global__ void hyp_finalize(const float4* __restrict__ partials, float* __restrict__ out) {
  __shared__ float s_sum[256];
  __shared__ float s_nv[256];
  float sum = 0.f, nv = 0.f;
  for (int r = threadIdx.x; r < N_TOT; r += 256) {
    float4 loc[N_SPLIT];
#pragma unroll
    for (int k = 0; k < N_SPLIT; ++k) loc[k] = partials[(size_t)r * N_SPLIT + k];
    float m = -INFINITY, S = 0.f, P = 0.f, np = 0.f;
#pragma unroll
    for (int k = 0; k < N_SPLIT; ++k) {
      m = fmaxf(m, loc[k].x);
      S += loc[k].y; P += loc[k].z; np += loc[k].w;
    }
    const float den = S * __expf(-m) + 1e-8f;
    const float row_sum = (P - np * m) - np * __logf(den);
    if (np > 0.f) { sum += row_sum / fmaxf(np, 1.f); nv += 1.f; }
  }
  s_sum[threadIdx.x] = sum; s_nv[threadIdx.x] = nv;
  __syncthreads();
  for (int s = 128; s > 0; s >>= 1) {
    if ((int)threadIdx.x < s) {
      s_sum[threadIdx.x] += s_sum[threadIdx.x + s];
      s_nv[threadIdx.x]  += s_nv[threadIdx.x + s];
    }
    __syncthreads();
  }
  if (threadIdx.x == 0) {
    const float nvv = s_nv[0];
    float loss = (nvv > 0.f) ? -(s_sum[0] / fmaxf(nvv, 1.f)) : 0.f;
    loss *= 0.5f;  // SCALE_BY_TEMPERATURE
    if (!__builtin_isfinite(loss)) loss = 0.f;
    out[0] = loss;
  }
}

// ------------------------------------------------------------------
extern "C" void kernel_launch(void* const* d_in, const int* in_sizes, int n_in,
                              void* d_out, int out_size, void* d_ws, size_t ws_size,
                              hipStream_t stream) {
  const float* features = (const float*)d_in[0];
  const int*   pl       = (const int*)d_in[1];
  const int*   sl       = (const int*)d_in[2];

  char* ws = (char*)d_ws;
  __bf16* hi = (__bf16*)ws;                                   // 8192*128*2 = 2 MB
  __bf16* lo = (__bf16*)(ws + (size_t)2 * 1024 * 1024);       // 2 MB
  float*  n2 = (float*)(ws + (size_t)4 * 1024 * 1024);        // 32 KB
  float4* partials = (float4*)(ws + (size_t)4 * 1024 * 1024 + 64 * 1024); // 2 MB

  (void)in_sizes; (void)n_in; (void)out_size; (void)ws_size;

  hyp_normalize<<<N_TOT / 8, 256, 0, stream>>>(features, hi, lo, n2);
  hyp_main<<<(N_TOT / 16) * N_SPLIT / 8, 256, 0, stream>>>(hi, lo, n2, pl, sl, partials);
  hyp_finalize<<<1, 256, 0, stream>>>(partials, (float*)d_out);
}
